// knn_cls_model_2894807957815
// MI455X (gfx1250) — matrix operand order repaired and checked by reading, not compiled, not run on hardware
//
#include <hip/hip_runtime.h>

// DGCNN-style KNN classifier for MI455X (gfx1250, wave32, WMMA).
// Pipeline: 4x [knn -> edgeconv(WMMA f16, fused BN+LeakyReLU+max-over-k)]
//           -> pointwise conv W5 (WMMA) + tile max -> global max reduce -> FC head.
//
// WMMA fragments are fed by contiguous ds_load_b128 pairs:
//  A (16x32 f16): lane(hf,m) holds K runs [hf*8, hf*8+8) and [16+hf*8, ...+8)
//  B (32x16 f16): lane(hf,n) holds K runs [hf*8, hf*8+8) and [16+hf*8, ...+8)
// Edge features staged K-contiguous per row; W tiles stored [n][K].
// B fragments are kk-invariant and hoisted into registers (KC x v16h),
// so the hot loop is 2x ds_load_b128 + 1x v_wmma per K-chunk.

typedef _Float16 f16t;
typedef __attribute__((ext_vector_type(16))) _Float16 v16h;
typedef __attribute__((ext_vector_type(8)))  _Float16 v8h;
typedef __attribute__((ext_vector_type(8)))  float    v8f;

#define KNN_K 10
#define NEG_SLOPE 0.2f
#define BATCH 8
#define NPTS 2048
#define EC_BLK 64
#define KNN_BLK 128

__device__ __forceinline__ float lrelu(float h) { return h >= 0.f ? h : NEG_SLOPE * h; }

__device__ __forceinline__ v16h cat16(v8h a, v8h b) {
    return __builtin_shufflevector(a, b, 0, 1, 2, 3, 4, 5, 6, 7,
                                   8, 9, 10, 11, 12, 13, 14, 15);
}

// ---------------------------------------------------------------------------
// KNN: one 128-thread block per (b, n). Per-thread sorted top-10 over strided
// candidates, then LDS k-way merge (10 argmin reductions). C is compile-time.
// ---------------------------------------------------------------------------
template <int C>
__global__ void knn_kernel(const float* __restrict__ x, int xStride, int xOff,
                           int N, int* __restrict__ idxOut)
{
    const int bn = blockIdx.x;
    const int b = bn / N, n = bn % N;
    const int tid = threadIdx.x; // KNN_BLK threads

    constexpr int CPad = (C + 3) & ~3;
    extern __shared__ char knnsm[];
    float* xi = (float*)knnsm;                   // CPad (16B aligned)
    float* cd = xi + CPad;                       // 128*10 candidate dists
    int*   ci = (int*)(cd + KNN_BLK * KNN_K);    // 128*10 candidate ids
    float* rd = (float*)(ci + KNN_BLK * KNN_K);  // 128 reduce dist
    int*   rc = (int*)(rd + KNN_BLK);            // 128 reduce candidate
    int*   rt = rc + KNN_BLK;                    // 128 reduce thread id
    int*   win = rt + KNN_BLK;                   // winner broadcast

    const float* xrow = x + ((size_t)b * N + n) * xStride + xOff;
    for (int c = tid; c < C; c += KNN_BLK) xi[c] = xrow[c];
    __syncthreads();

    float dloc[KNN_K]; int iloc[KNN_K];
    #pragma unroll
    for (int q = 0; q < KNN_K; ++q) { dloc[q] = 3.4e38f; iloc[q] = 0; }

    for (int j = tid; j < N; j += KNN_BLK) {
        if (j == n) continue;                // exclude self (reference drops it)
        const float* xj = x + ((size_t)b * N + j) * xStride + xOff;
        float d = 0.f;
        if ((C & 3) == 0) {
            const float4* xj4 = (const float4*)xj;
            const float4* xi4 = (const float4*)xi;
            #pragma unroll
            for (int c4 = 0; c4 < (C >> 2); ++c4) {
                float4 a = xj4[c4], bv = xi4[c4];
                float d0 = a.x - bv.x, d1 = a.y - bv.y;
                float d2 = a.z - bv.z, d3 = a.w - bv.w;
                d += d0 * d0 + d1 * d1 + d2 * d2 + d3 * d3;
            }
        } else {
            #pragma unroll
            for (int c = 0; c < C; ++c) { float df = xj[c] - xi[c]; d += df * df; }
        }
        if (d < dloc[KNN_K - 1]) {
            int p = KNN_K - 1;
            while (p > 0 && dloc[p - 1] > d) {
                dloc[p] = dloc[p - 1]; iloc[p] = iloc[p - 1]; --p;
            }
            dloc[p] = d; iloc[p] = j;
        }
    }
    #pragma unroll
    for (int q = 0; q < KNN_K; ++q) {
        cd[tid * KNN_K + q] = dloc[q];
        ci[tid * KNN_K + q] = iloc[q];
    }
    int hp = 0;
    __syncthreads();

    for (int sel = 0; sel < KNN_K; ++sel) {
        rd[tid] = (hp < KNN_K) ? cd[tid * KNN_K + hp] : 3.4e38f;
        rc[tid] = (hp < KNN_K) ? ci[tid * KNN_K + hp] : 0;
        rt[tid] = tid;
        __syncthreads();
        for (int s = KNN_BLK / 2; s > 0; s >>= 1) {
            if (tid < s && rd[tid + s] < rd[tid]) {
                rd[tid] = rd[tid + s]; rc[tid] = rc[tid + s]; rt[tid] = rt[tid + s];
            }
            __syncthreads();
        }
        if (tid == 0) {
            idxOut[((size_t)b * N + n) * KNN_K + sel] = rc[0];
            win[0] = rt[0];
        }
        __syncthreads();
        if (tid == win[0]) hp++;
        __syncthreads();
    }
}

// ---------------------------------------------------------------------------
// EdgeConv via WMMA f32<=f16 16x16x32. Block = 64 threads (2 waves).
// M-tile = 16 points, N-tile = 16 out channels; neighbor kk is an outer loop
// with the max-over-neighbors fused into the f32 epilogue.
// LDS: edgeA[10][16][KPad] f16 (K-contiguous edge features [nbr-xi, xi]),
//      per-wave W tile [16][KPad] f16, xi[16][CIN] f32, nb[16][10] int.
// ---------------------------------------------------------------------------
template <int CIN, int COUT>
__global__ void edgeconv_kernel(const float* __restrict__ x, int xStride, int xOff,
                                const int* __restrict__ nnIdx,
                                const float* __restrict__ W,
                                const float* __restrict__ s, const float* __restrict__ t,
                                float* __restrict__ out, int outStride, int outOff,
                                int N)
{
    constexpr int K2 = 2 * CIN;
    constexpr int KC = (K2 + 31) / 32;
    constexpr int KPad = KC * 32;
    constexpr int NUMOT = COUT / 16;
    constexpr int ITERS = NUMOT / 2;      // 2 waves, uniform loop count

    const int ptile = blockIdx.x, b = blockIdx.y;
    const int ptBase = ptile * 16;
    const int tid = threadIdx.x;          // EC_BLK
    const int wave = tid >> 5, lane = tid & 31;

    extern __shared__ char smraw[];
    f16t*  edgeA = (f16t*)smraw;                          // [10][16][KPad]
    f16t*  wl    = edgeA + (size_t)KNN_K * 16 * KPad;     // [2][16][KPad]
    f16t*  wtile = wl + (size_t)wave * 16 * KPad;
    float* xi    = (float*)(wl + (size_t)2 * 16 * KPad);  // [16][CIN] f32
    int*   nb    = (int*)(xi + 16 * CIN);                 // [16][10]

    // Stage center features (f32) and neighbor ids.
    for (int m = 0; m < 16; ++m)
        for (int c = tid; c < CIN; c += EC_BLK)
            xi[m * CIN + c] = x[((size_t)b * N + ptBase + m) * xStride + xOff + c];
    for (int i = tid; i < 16 * KNN_K; i += EC_BLK) {
        int m = i / KNN_K, q = i - m * KNN_K;
        nb[i] = nnIdx[((size_t)b * N + ptBase + m) * KNN_K + q];
    }
    __syncthreads();

    // Build edge features [nbr - xi, xi], zero-padded to KPad, K-contiguous.
    for (int kk = 0; kk < KNN_K; ++kk) {
        for (int m = 0; m < 16; ++m) {
            const int j = nb[m * KNN_K + kk];
            const float* xj = x + ((size_t)b * N + j) * xStride + xOff;
            f16t* erow = edgeA + ((size_t)kk * 16 + m) * KPad;
            for (int k = tid; k < KPad; k += EC_BLK) {
                float v;
                if (k < CIN)      v = xj[k] - xi[m * CIN + k];
                else if (k < K2)  v = xi[m * CIN + (k - CIN)];
                else              v = 0.f;
                erow[k] = (f16t)v;
            }
        }
    }
    __syncthreads();

    const int hf = lane >> 4;
    const int nl = lane & 15;

    for (int it = 0; it < ITERS; ++it) {
        const int oBase = (wave + it * 2) * 16;

        // Per-wave W tile, stored [n][KPad] so B fragments are K-contiguous.
        __builtin_prefetch(W + (size_t)oBase * K2, 0, 0);
        for (int n = 0; n < 16; ++n) {
            const float* wrow = W + (size_t)(oBase + n) * K2;
            f16t* drow = wtile + n * KPad;
            for (int k = lane; k < KPad; k += 32)
                drow[k] = (k < K2) ? (f16t)wrow[k] : (f16t)0.f;
        }
        __syncthreads();

        // Hoist kk-invariant B fragments into registers.
        const f16t* brow = wtile + (size_t)nl * KPad;
        v16h bfr[KC];
        #pragma unroll
        for (int kc = 0; kc < KC; ++kc) {
            v8h b0 = *(const v8h*)(brow + kc * 32 + hf * 8);
            v8h b1 = *(const v8h*)(brow + kc * 32 + 16 + hf * 8);
            bfr[kc] = cat16(b0, b1);
        }

        const float sv = s[oBase + nl];
        const float tv = t[oBase + nl];
        float rmax[8];
        #pragma unroll
        for (int r = 0; r < 8; ++r) rmax[r] = -3.4e38f;

        for (int kk = 0; kk < KNN_K; ++kk) {
            const f16t* arow = edgeA + ((size_t)kk * 16 + nl) * KPad;
            v8f acc = {};
            #pragma unroll
            for (int kc = 0; kc < KC; ++kc) {
                v8h a0 = *(const v8h*)(arow + kc * 32 + hf * 8);
                v8h a1 = *(const v8h*)(arow + kc * 32 + 16 + hf * 8);
                acc = __builtin_amdgcn_wmma_f32_16x16x32_f16(
                          false, cat16(a0, a1), false, bfr[kc],
                          (short)0, acc, false, false);
            }
            #pragma unroll
            for (int r = 0; r < 8; ++r) {
                float h = lrelu(acc[r] * sv + tv);
                rmax[r] = fmaxf(rmax[r], h);
            }
        }
        // Lane holds column nl, rows r + 8*hf.
        #pragma unroll
        for (int r = 0; r < 8; ++r) {
            int m = r + hf * 8;
            out[((size_t)b * N + ptBase + m) * outStride + outOff + oBase + nl] = rmax[r];
        }
        __syncthreads();
    }
}

// ---------------------------------------------------------------------------
// Pointwise conv W5 (512 -> 1024) via WMMA + per-tile max (partial pooling).
// grid (N/16, 32, B), block 64 (2 waves); wave handles out-tile g*2+wave.
// ---------------------------------------------------------------------------
__global__ void conv5_kernel(const float* __restrict__ xcat,
                             const float* __restrict__ W5,
                             const float* __restrict__ s5, const float* __restrict__ t5,
                             float* __restrict__ partial, int N)
{
    constexpr int Cin = 512, KC = 16;
    const int ptile = blockIdx.x, g = blockIdx.y, b = blockIdx.z;
    const int ptBase = ptile * 16;
    const int tid = threadIdx.x, wave = tid >> 5, lane = tid & 31;

    extern __shared__ char smraw5[];
    f16t* feat  = (f16t*)smraw5;                // [16][512]
    f16t* wl    = feat + 16 * Cin;              // [2][16][512]
    f16t* wtile = wl + (size_t)wave * 16 * Cin;

    for (int m = 0; m < 16; ++m) {
        const float* xr = xcat + ((size_t)b * N + ptBase + m) * 512;
        for (int c = tid; c < Cin; c += EC_BLK)
            feat[m * Cin + c] = (f16t)xr[c];
    }
    const int ot = g * 2 + wave, oBase = ot * 16;
    __builtin_prefetch(W5 + (size_t)oBase * Cin, 0, 0);
    for (int n = 0; n < 16; ++n) {
        const float* wrow = W5 + (size_t)(oBase + n) * Cin;
        f16t* drow = wtile + n * Cin;
        for (int k = lane; k < Cin; k += 32) drow[k] = (f16t)wrow[k];
    }
    __syncthreads();

    const int hf = lane >> 4, nl = lane & 15;
    const f16t* arow = feat + (size_t)nl * Cin;
    const f16t* brow = wtile + (size_t)nl * Cin;
    v8f acc = {};
    #pragma unroll
    for (int kc = 0; kc < KC; ++kc) {
        v8h a0 = *(const v8h*)(arow + kc * 32 + hf * 8);
        v8h a1 = *(const v8h*)(arow + kc * 32 + 16 + hf * 8);
        v8h b0 = *(const v8h*)(brow + kc * 32 + hf * 8);
        v8h b1 = *(const v8h*)(brow + kc * 32 + 16 + hf * 8);
        acc = __builtin_amdgcn_wmma_f32_16x16x32_f16(
                  false, cat16(a0, a1), false, cat16(b0, b1),
                  (short)0, acc, false, false);
    }
    const float sv = s5[oBase + nl], tv = t5[oBase + nl];
    float mx = -3.4e38f;
    #pragma unroll
    for (int r = 0; r < 8; ++r) mx = fmaxf(mx, lrelu(acc[r] * sv + tv));
    mx = fmaxf(mx, __shfl_xor(mx, 16, 32));       // combine the two row halves
    if (hf == 0)
        partial[(((size_t)b * (N / 16) + ptile) * 1024) + oBase + nl] = mx;
}

// pooled[b][o] = max over point tiles of partial.
__global__ void pool_reduce_kernel(const float* __restrict__ partial,
                                   float* __restrict__ pooled, int tiles)
{
    int i = blockIdx.x * blockDim.x + threadIdx.x;
    if (i >= BATCH * 1024) return;
    int b = i >> 10, o = i & 1023;
    float mx = -3.4e38f;
    for (int p = 0; p < tiles; ++p)
        mx = fmaxf(mx, partial[(((size_t)b * tiles + p) * 1024) + o]);
    pooled[i] = mx;
}

// FC head: tiny (8 rows), scalar. One block per batch element.
__global__ void fc_head_kernel(const float* __restrict__ pooled,
                               const float* __restrict__ Wf1, const float* __restrict__ sf1,
                               const float* __restrict__ tf1,
                               const float* __restrict__ Wf2, const float* __restrict__ bf2,
                               const float* __restrict__ sf2, const float* __restrict__ tf2,
                               const float* __restrict__ Wf3, const float* __restrict__ bf3,
                               float* __restrict__ outp)
{
    const int b = blockIdx.x, tid = threadIdx.x; // 512 threads
    __shared__ float h6[512];
    __shared__ float h7[256];
    const float* pb = pooled + (size_t)b * 1024;

    if (tid < 512) {
        float a = 0.f;
        for (int c = 0; c < 1024; ++c) a += pb[c] * Wf1[(size_t)tid * 1024 + c];
        h6[tid] = lrelu(a * sf1[tid] + tf1[tid]);
    }
    __syncthreads();
    if (tid < 256) {
        float a = 0.f;
        for (int c = 0; c < 512; ++c) a += h6[c] * Wf2[(size_t)tid * 512 + c];
        h7[tid] = lrelu((a + bf2[tid]) * sf2[tid] + tf2[tid]);
    }
    __syncthreads();
    if (tid < 3) {
        float a = 0.f;
        for (int c = 0; c < 256; ++c) a += h7[c] * Wf3[(size_t)tid * 256 + c];
        outp[b * 3 + tid] = a + bf3[tid];
    }
}

// ---------------------------------------------------------------------------
extern "C" void kernel_launch(void* const* d_in, const int* in_sizes, int n_in,
                              void* d_out, int out_size, void* d_ws, size_t ws_size,
                              hipStream_t stream)
{
    (void)in_sizes; (void)n_in; (void)out_size; (void)ws_size;
    const float* points = (const float*)d_in[0];
    const float* W1 = (const float*)d_in[1];  const float* s1 = (const float*)d_in[2];  const float* t1 = (const float*)d_in[3];
    const float* W2 = (const float*)d_in[4];  const float* s2 = (const float*)d_in[5];  const float* t2 = (const float*)d_in[6];
    const float* W3 = (const float*)d_in[7];  const float* s3 = (const float*)d_in[8];  const float* t3 = (const float*)d_in[9];
    const float* W4 = (const float*)d_in[10]; const float* s4 = (const float*)d_in[11]; const float* t4 = (const float*)d_in[12];
    const float* W5 = (const float*)d_in[13]; const float* s5 = (const float*)d_in[14]; const float* t5 = (const float*)d_in[15];
    const float* Wf1 = (const float*)d_in[16]; const float* sf1 = (const float*)d_in[17]; const float* tf1 = (const float*)d_in[18];
    const float* Wf2 = (const float*)d_in[19]; const float* bf2 = (const float*)d_in[20];
    const float* sf2 = (const float*)d_in[21]; const float* tf2 = (const float*)d_in[22];
    const float* Wf3 = (const float*)d_in[23]; const float* bf3 = (const float*)d_in[24];

    const int B = BATCH, N = NPTS;
    const int TILES = N / 16; // 128

    // Workspace layout (256B aligned slices).
    char* ws = (char*)d_ws;
    size_t off = 0;
    auto take = [&](size_t bytes) -> char* {
        char* p = ws + off;
        off = (off + bytes + 255) & ~(size_t)255;
        return p;
    };
    int*   idxBuf  = (int*)  take((size_t)B * N * KNN_K * sizeof(int));
    float* catbuf  = (float*)take((size_t)B * N * 512 * sizeof(float));
    float* partial = (float*)take((size_t)B * TILES * 1024 * sizeof(float));
    float* pooled  = (float*)take((size_t)B * 1024 * sizeof(float));

    auto knnSh = [](int C) -> size_t {
        return (size_t)((C + 3) & ~3) * 4 + KNN_BLK * KNN_K * 8 + KNN_BLK * 12 + 16;
    };
    auto ecSh = [](int Cin) -> size_t {
        size_t KPad = (size_t)(((2 * Cin + 31) / 32) * 32);
        return (KNN_K * 16 * KPad + 2 * 16 * KPad) * sizeof(f16t)  // edgeA + W tiles
             + (size_t)16 * Cin * sizeof(float)                    // xi
             + (size_t)16 * KNN_K * sizeof(int);                   // nb
    };

    dim3 ecGrid(TILES, B);

    // EdgeConv 1: points(3) -> cat[:, :, 0:64]
    knn_kernel<3><<<B * N, KNN_BLK, knnSh(3), stream>>>(points, 3, 0, N, idxBuf);
    edgeconv_kernel<3, 64><<<ecGrid, EC_BLK, ecSh(3), stream>>>(points, 3, 0, idxBuf,
        W1, s1, t1, catbuf, 512, 0, N);

    // EdgeConv 2: cat[0:64] -> cat[64:128]
    knn_kernel<64><<<B * N, KNN_BLK, knnSh(64), stream>>>(catbuf, 512, 0, N, idxBuf);
    edgeconv_kernel<64, 64><<<ecGrid, EC_BLK, ecSh(64), stream>>>(catbuf, 512, 0, idxBuf,
        W2, s2, t2, catbuf, 512, 64, N);

    // EdgeConv 3: cat[64:128] -> cat[128:256]
    knn_kernel<64><<<B * N, KNN_BLK, knnSh(64), stream>>>(catbuf, 512, 64, N, idxBuf);
    edgeconv_kernel<64, 128><<<ecGrid, EC_BLK, ecSh(64), stream>>>(catbuf, 512, 64, idxBuf,
        W3, s3, t3, catbuf, 512, 128, N);

    // EdgeConv 4: cat[128:256] -> cat[256:512]
    knn_kernel<128><<<B * N, KNN_BLK, knnSh(128), stream>>>(catbuf, 512, 128, N, idxBuf);
    edgeconv_kernel<128, 256><<<ecGrid, EC_BLK, ecSh(128), stream>>>(catbuf, 512, 128, idxBuf,
        W4, s4, t4, catbuf, 512, 256, N);

    // Pointwise conv W5 + tile-level max pooling.
    size_t sh5 = ((size_t)16 * 512 + 2 * (size_t)16 * 512) * sizeof(f16t);
    conv5_kernel<<<dim3(TILES, 32, B), EC_BLK, sh5, stream>>>(catbuf, W5, s5, t5, partial, N);
    pool_reduce_kernel<<<(B * 1024 + 255) / 256, 256, 0, stream>>>(partial, pooled, TILES);

    // FC head -> d_out (B x 3).
    fc_head_kernel<<<B, 512, 0, stream>>>(pooled, Wf1, sf1, tf1,
                                          Wf2, bf2, sf2, tf2, Wf3, bf3,
                                          (float*)d_out);
}
